// GINPlusModel_67345087201312
// MI455X (gfx1250) — matrix operand order repaired and checked by reading, not compiled, not run on hardware
//
#include <hip/hip_runtime.h>
#include <cmath>
#include <cstdint>

// ---------------------------------------------------------------------------
// MI455X (gfx1250) GIN+ forward.
//  * One templated WMMA GEMM (bf16 in, f32 acc, v_wmma_f32_16x16x32_bf16).
//    Block tile 128x64, 8 wave32 waves x (32x32) = 4 WMMA/wave/K-step.
//    B operand pre-transposed so all global/LDS traffic is 128-bit.
//  * Gathers (edge endpoints, vn broadcast) fused into the A-tile loader,
//    hoisted out of the K loop. Softmax / scatter / pooling via f32 atomics.
// ---------------------------------------------------------------------------

typedef __bf16 bf16_t;
typedef __attribute__((ext_vector_type(16))) __bf16 v16bf;
typedef __attribute__((ext_vector_type(8)))  float  v8f;

#define NN    50000
#define NNP   50048     // padded nodes (multiple of 128)
#define NED   250000
#define NEDP  250112    // padded edges (multiple of 128)
#define NG    512
#define HP    320       // padded H (300)
#define H2P   640       // padded 2H (600)
#define P3P   960       // padded 3H (900)

#define FLAG_RELU 1
#define FLAG_TANH 2

#define BN_SCALE_C 0.9999950000374997f   // 1/sqrt(1+1e-5)

__device__ __forceinline__ float bf2f(bf16_t v) { return (float)v; }

__device__ __forceinline__ void atomicMaxF(float* addr, float v) {
  if (v >= 0.f) atomicMax((int*)addr, __float_as_int(v));
  else          atomicMin((unsigned int*)addr, (unsigned int)__float_as_int(v));
}

// ----------------------------- GEMM ----------------------------------------
struct GemmA {
  const bf16_t* a0; const bf16_t* addb; const int* idx0;   // first K segment
  const bf16_t* a1; const int* idx1;                       // second K segment
  int lda0, lda1, ksplit, Mreal;
};
struct GemmEpi {
  const float* bias; const float* bng; const float* bnb; const float* add0;
  float* outF; bf16_t* outB;
  int ldo, ldadd, Nreal, Nstore, flags;
};

// MODE: 0 = direct A ; 1 = concat [a0 | a1[idx1]] ; 2 = concat [a0[idx0]+addb | a1[idx1]]
// Wt is the weight transposed: Wt[n * Kp + k]. ksplit is a multiple of 32.
template <int MODE>
__global__ __launch_bounds__(256) void k_gemm(GemmA A, const bf16_t* __restrict__ Wt,
                                              int Kp, GemmEpi E)
{
  __shared__ bf16_t sA[128][40];   // 80B row stride: 16B aligned, conflict-friendly
  __shared__ bf16_t sBt[64][40];   // B tile stored K-contiguous per output column

  const int tid  = threadIdx.x;
  const int wv   = tid >> 5, lane = tid & 31;
  const int wm   = (wv & 3) * 32;   // 0..96
  const int wn   = (wv >> 2) * 32;  // 0,32
  const int bm   = blockIdx.y * 128;
  const int bn   = blockIdx.x * 64;

  // A staging: thread covers rows {ar, ar+64}, cols [ac, ac+8)
  const int ar = tid >> 2;
  const int ac = (tid & 3) * 8;
  // B staging: thread covers column bnr, k [bkc, bkc+8)
  const int bnr = tid >> 2;
  const int bkc = (tid & 3) * 8;

  // hoisted per-row descriptors (gather indices read once, not per K element)
  const bf16_t* rp0[2]; const bf16_t* rpa[2]; const bf16_t* rp1[2]; bool rv[2];
  #pragma unroll
  for (int t = 0; t < 2; ++t) {
    const int gm  = bm + ar + t * 64;
    const bool v  = gm < A.Mreal;
    const int gmc = v ? gm : 0;
    rv[t] = v;
    if (MODE == 0) {
      rp0[t] = A.a0 + (size_t)gmc * A.lda0;
      rpa[t] = rp0[t]; rp1[t] = rp0[t];
    } else if (MODE == 1) {
      rp0[t] = A.a0 + (size_t)gmc * A.lda0;
      rp1[t] = A.a1 + (size_t)A.idx1[gmc] * A.lda1;
      rpa[t] = rp0[t];
    } else {
      rp0[t] = A.a0 + (size_t)A.idx0[gmc] * A.lda0;
      rpa[t] = A.addb + (size_t)gmc * A.lda0;
      rp1[t] = A.a1 + (size_t)A.idx1[gmc] * A.lda1;
    }
  }
  const bf16_t* wrow = Wt + (size_t)(bn + bnr) * Kp;

  const v8f vz = {0.f,0.f,0.f,0.f,0.f,0.f,0.f,0.f};
  v8f acc[2][2] = {{vz, vz}, {vz, vz}};

  for (int kt = 0; kt < Kp; kt += 32) {
    // ---- stage A tile (128 x 32), 16B vector loads ----
    #pragma unroll
    for (int t = 0; t < 2; ++t) {
      const int k = kt + ac;
      uint4 q; q.x = 0u; q.y = 0u; q.z = 0u; q.w = 0u;
      if (rv[t]) {
        if (MODE == 0) {
          q = *(const uint4*)(rp0[t] + k);
        } else if (MODE == 1) {
          q = (k < A.ksplit) ? *(const uint4*)(rp0[t] + k)
                             : *(const uint4*)(rp1[t] + (k - A.ksplit));
        } else {
          if (k < A.ksplit) {
            uint4 qa = *(const uint4*)(rp0[t] + k);
            uint4 qb = *(const uint4*)(rpa[t] + k);
            bf16_t* xa = (bf16_t*)&qa; const bf16_t* xb = (const bf16_t*)&qb;
            #pragma unroll
            for (int j = 0; j < 8; ++j) xa[j] = (bf16_t)((float)xa[j] + (float)xb[j]);
            q = qa;
          } else {
            q = *(const uint4*)(rp1[t] + (k - A.ksplit));
          }
        }
      }
      *(uint4*)&sA[ar + t * 64][ac] = q;
    }
    // ---- stage B tile (64 cols x 32 k), K-contiguous: 16B load + 16B LDS store
    *(uint4*)&sBt[bnr][bkc] = *(const uint4*)(wrow + kt + bkc);
    __syncthreads();

    // ---- fragments (two ds_load_b128 each) ----
    v16bf af[2], bfr[2];
    const int abase = (lane < 16) ? 0 : 8;
    #pragma unroll
    for (int t = 0; t < 2; ++t) {
      const int mrow = wm + t * 16 + (lane & 15);
      union { uint4 q[2]; v16bf v; } u;
      u.q[0] = *(const uint4*)&sA[mrow][abase];
      u.q[1] = *(const uint4*)&sA[mrow][abase + 16];
      af[t] = u.v;
    }
    const int bbase = (lane < 16) ? 0 : 8;
    #pragma unroll
    for (int s = 0; s < 2; ++s) {
      const int ncol = wn + s * 16 + (lane & 15);
      union { uint4 q[2]; v16bf v; } u;
      u.q[0] = *(const uint4*)&sBt[ncol][bbase];
      u.q[1] = *(const uint4*)&sBt[ncol][bbase + 16];
      bfr[s] = u.v;
    }
    #pragma unroll
    for (int t = 0; t < 2; ++t)
      #pragma unroll
      for (int s = 0; s < 2; ++s)
        acc[t][s] = __builtin_amdgcn_wmma_f32_16x16x32_bf16(false, af[t], false, bfr[s],
                                                            (short)0, acc[t][s], false, false);
    __syncthreads();
  }

  // ---- epilogue: bias + BN + residual + act, dual f32/bf16 stores ----
  #pragma unroll
  for (int t = 0; t < 2; ++t) {
    const int mb = bm + wm + t * 16 + ((lane < 16) ? 0 : 8);
    #pragma unroll
    for (int s = 0; s < 2; ++s) {
      const int gn = bn + wn + s * 16 + (lane & 15);
      #pragma unroll
      for (int r = 0; r < 8; ++r) {
        const int gm = mb + r;
        float v = 0.f;
        if (gm < A.Mreal && gn < E.Nreal) {
          v = acc[t][s][r];
          if (E.bias) v += E.bias[gn];
          if (E.bng)  v = E.bng[gn] * (v * BN_SCALE_C) + E.bnb[gn];
          if (E.add0) v += E.add0[(size_t)gm * E.ldadd + gn];
          if (E.flags & FLAG_RELU) v = fmaxf(v, 0.f);
          if (E.flags & FLAG_TANH) v = tanhf(v);
        }
        if (gn < E.Nstore) {
          if (E.outF) E.outF[(size_t)gm * E.ldo + gn] = v;
          if (E.outB) E.outB[(size_t)gm * E.ldo + gn] = (bf16_t)v;
        }
      }
    }
  }
}

// ------------------------- small utility kernels ---------------------------
__global__ void k_zero_f(float* p, int n) {
  int i = blockIdx.x * 256 + threadIdx.x; if (i < n) p[i] = 0.f;
}
__global__ void k_fill_f(float* p, int n, float v) {
  int i = blockIdx.x * 256 + threadIdx.x; if (i < n) p[i] = v;
}
__global__ void k_zero_b(bf16_t* p, int n) {
  int i = blockIdx.x * 256 + threadIdx.x; if (i < n) p[i] = (bf16_t)0.f;
}
// plain activation convert (row-major, zero-padded)
__global__ void k_convert_pad(const float* __restrict__ src, bf16_t* __restrict__ dst,
                              int K, int N, int Kp, int Np) {
  int i = blockIdx.x * 256 + threadIdx.x; if (i >= Kp * Np) return;
  int k = i / Np, nn = i % Np;
  float v = (k < K && nn < N) ? src[(size_t)k * N + nn] : 0.f;
  dst[i] = (bf16_t)v;
}
// weight repack TRANSPOSED: dst[n*Kp + k] = src[k*N + n], zero-padded
__global__ void k_convert_wT(const float* __restrict__ src, bf16_t* __restrict__ dst,
                             int K, int N, int Kp, int Np) {
  int i = blockIdx.x * 256 + threadIdx.x; if (i >= Np * Kp) return;
  int n = i / Kp, k = i % Kp;
  float v = (k < K && n < N) ? src[(size_t)k * N + n] : 0.f;
  dst[i] = (bf16_t)v;
}
// transposed repack for concat-K GEMMs: k rows [0,K1) then [S,S+K2) <- src[K1..]
__global__ void k_convert_catT(const float* __restrict__ src, bf16_t* __restrict__ dst,
                               int K1, int S, int K2, int N, int Kp, int Np) {
  int i = blockIdx.x * 256 + threadIdx.x; if (i >= Np * Kp) return;
  int n = i / Kp, k = i % Kp;
  float v = 0.f;
  if (n < N) {
    if (k < K1)                    v = src[(size_t)k * N + n];
    else if (k >= S && k < S + K2) v = src[(size_t)(K1 + k - S) * N + n];
  }
  dst[i] = (bf16_t)v;
}
__global__ void k_count(const int* __restrict__ batch, float* cnt) {
  int i = blockIdx.x * 256 + threadIdx.x; if (i < NN) atomicAdd(&cnt[batch[i]], 1.f);
}
__global__ void k_segsum(const float* __restrict__ h, const int* __restrict__ batch,
                         float* gsum) {
  int i = blockIdx.x * 256 + threadIdx.x;
  if (i >= NN * HP) return;
  int n = i / HP, k = i % HP;
  atomicAdd(&gsum[(size_t)batch[n] * HP + k], h[i]);
}
__global__ void k_vn_prep(const float* __restrict__ gsum, const float* __restrict__ cnt,
                          const float* __restrict__ emb, bf16_t* vinb) {
  int i = blockIdx.x * 256 + threadIdx.x; if (i >= NG * HP) return;
  int g = i / HP, k = i % HP;
  float v = (k < 300) ? (emb[k] + gsum[i] / fmaxf(cnt[g], 1.f)) : 0.f;
  vinb[i] = (bf16_t)v;
}
// per-row dot with small weight vector (+bias, leaky-relu); pad rows -> -1e30
__global__ __launch_bounds__(256) void k_dot_reduce(const bf16_t* __restrict__ t, int ld,
                                                    int Kreal, const float* __restrict__ w,
                                                    const float* __restrict__ b, float slope,
                                                    int Mreal, float* out) {
  int row = blockIdx.x * 8 + (threadIdx.x >> 5);
  int lane = threadIdx.x & 31;
  float s = 0.f;
  for (int k = lane; k < Kreal; k += 32) s += bf2f(t[(size_t)row * ld + k]) * w[k];
  for (int o = 16; o > 0; o >>= 1) s += __shfl_down(s, o, 32);
  if (lane == 0) {
    float v = s + b[0];
    v = (v > 0.f) ? v : slope * v;
    out[row] = (row < Mreal) ? v : -1e30f;
  }
}
__global__ void k_scr_init(float* scr) {
  if (threadIdx.x == 0 && blockIdx.x == 0) { scr[0] = -1e30f; scr[1] = 0.f; }
}
__global__ __launch_bounds__(256) void k_amax(const float* __restrict__ v, int n, float* scr) {
  __shared__ float sm[256];
  int i = blockIdx.x * 256 + threadIdx.x;
  sm[threadIdx.x] = (i < n) ? v[i] : -1e30f;
  __syncthreads();
  for (int o = 128; o > 0; o >>= 1) {
    if (threadIdx.x < o) sm[threadIdx.x] = fmaxf(sm[threadIdx.x], sm[threadIdx.x + o]);
    __syncthreads();
  }
  if (threadIdx.x == 0) atomicMaxF(&scr[0], sm[0]);
}
__global__ __launch_bounds__(256) void k_expsum(float* v, int n, float* scr) {
  __shared__ float sm[256];
  int i = blockIdx.x * 256 + threadIdx.x;
  float e = 0.f, mx = scr[0];
  if (i < n) { e = expf(v[i] - mx); v[i] = e; }
  sm[threadIdx.x] = e;
  __syncthreads();
  for (int o = 128; o > 0; o >>= 1) {
    if (threadIdx.x < o) sm[threadIdx.x] += sm[threadIdx.x + o];
    __syncthreads();
  }
  if (threadIdx.x == 0) atomicAdd(&scr[1], sm[0]);
}
// agg[row[e]] += (xt[col[e]] + ef[e]) * softmax(a)[e]
__global__ __launch_bounds__(256) void k_scatter(const bf16_t* __restrict__ xtb,
                                                 const bf16_t* __restrict__ efb,
                                                 const int* __restrict__ row,
                                                 const int* __restrict__ col,
                                                 const float* __restrict__ ea,
                                                 const float* __restrict__ scr,
                                                 float* agg) {
  int e = blockIdx.x * 8 + (threadIdx.x >> 5);
  if (e >= NED) return;
  int lane = threadIdx.x & 31;
  float w = ea[e] / scr[1];
  int cs = col[e], rd = row[e];
  for (int k = lane; k < HP; k += 32) {
    float v = (bf2f(xtb[(size_t)cs * HP + k]) + bf2f(efb[(size_t)e * HP + k])) * w;
    atomicAdd(&agg[(size_t)rd * HP + k], v);
  }
}
__global__ void k_combine(const float* __restrict__ xt, const float* __restrict__ eps,
                          const float* __restrict__ agg, const float* __restrict__ vu,
                          const int* __restrict__ batch, const float* __restrict__ na,
                          const float* __restrict__ scrN, bf16_t* cmb) {
  int i = blockIdx.x * 256 + threadIdx.x;
  if (i >= NNP * HP) return;
  int n = i / HP, k = i % HP;
  float v = 0.f;
  if (n < NN) {
    float aw = na[n] / scrN[1];
    v = (1.f + eps[0]) * xt[i] + agg[i] + vu[(size_t)batch[n] * HP + k] * aw;
  }
  cmb[i] = (bf16_t)v;
}
__global__ void k_pool_acc(const float* __restrict__ h, const int* __restrict__ batch,
                           float* ps, float* pmx) {
  int i = blockIdx.x * 256 + threadIdx.x;
  if (i >= NN * HP) return;
  int n = i / HP, k = i % HP;
  int g = batch[n];
  atomicAdd(&ps[(size_t)g * HP + k], h[i]);
  atomicMaxF(&pmx[(size_t)g * HP + k], h[i]);
}
__global__ void k_pool_pack(const float* __restrict__ ps, const float* __restrict__ pmx,
                            const float* __restrict__ cnt, bf16_t* gbuf, float* gout) {
  int i = blockIdx.x * 256 + threadIdx.x; if (i >= NG * P3P) return;
  int g = i / P3P, k = i % P3P;
  float v = 0.f;
  if (k < 900) {
    int seg = k / 300, kk = k % 300;
    float s = ps[g * HP + kk];
    if (seg == 0)      v = s;
    else if (seg == 1) v = s / fmaxf(cnt[g], 1.f);
    else               v = (cnt[g] > 0.f) ? pmx[g * HP + kk] : 0.f;
    gout[g * 900 + k] = v;
  }
  gbuf[i] = (bf16_t)v;
}
__global__ void k_l3(const float* __restrict__ z2, const float* __restrict__ w,
                     const float* __restrict__ b, float* out) {
  int i = blockIdx.x * 256 + threadIdx.x; if (i >= NG * 6) return;
  int g = i / 6, o = i % 6;
  float s = b[o];
  for (int k = 0; k < 150; ++k) s += z2[g * 160 + k] * w[k * 6 + o];
  out[i] = s;
}
__global__ __launch_bounds__(256) void k_f2(const float* __restrict__ z3,
                                            const float* __restrict__ w,
                                            const float* __restrict__ b, float* out) {
  int g = blockIdx.x * 8 + (threadIdx.x >> 5);
  int lane = threadIdx.x & 31;
  if (g >= NG) return;
  float s = 0.f;
  for (int k = lane; k < 225; k += 32) s += z3[g * 256 + k] * w[k];
  for (int o = 16; o > 0; o >>= 1) s += __shfl_down(s, o, 32);
  if (lane == 0) out[g] = 1.f / (1.f + expf(-(s + b[0])));
}

// --------------------------------------------------------------------------
extern "C" void kernel_launch(void* const* d_in, const int* in_sizes, int n_in,
                              void* d_out, int out_size, void* d_ws, size_t ws_size,
                              hipStream_t stream) {
  (void)in_sizes; (void)n_in; (void)out_size; (void)ws_size;
  const float* x     = (const float*)d_in[0];
  const int*   ei    = (const int*)d_in[1];
  const float* eain  = (const float*)d_in[2];
  const int*   batch = (const int*)d_in[3];
  const int* row = ei;            // edge_index[0]
  const int* col = ei + NED;      // edge_index[1]

  auto P = [&](int i) -> const float* { return (const float*)d_in[i]; };
  const int PB = 4;
  auto GIN = [&](int l, int o) { return PB + 4 + 19 * l + o; };
  const int VNB = PB + 4 + 95, CLFB = VNB + 11, NFB = CLFB + 10;

  char* wsb = (char*)d_ws; size_t off = 0;
  auto alloc = [&](size_t bytes) -> void* {
    void* p = wsb + off; off += (bytes + 255) & ~(size_t)255; return p;
  };
  // transposed, padded bf16 weight repack (Np multiple of 64)
  auto convW = [&](const float* src, int K, int N, int Kp, int Np) -> bf16_t* {
    bf16_t* dst = (bf16_t*)alloc((size_t)Np * Kp * sizeof(bf16_t));
    int n = Np * Kp;
    k_convert_wT<<<(n + 255) / 256, 256, 0, stream>>>(src, dst, K, N, Kp, Np);
    return dst;
  };
  auto convCat = [&](const float* src, int K1, int S, int K2, int N, int Kp, int Np) -> bf16_t* {
    bf16_t* dst = (bf16_t*)alloc((size_t)Np * Kp * sizeof(bf16_t));
    int n = Np * Kp;
    k_convert_catT<<<(n + 255) / 256, 256, 0, stream>>>(src, dst, K1, S, K2, N, Kp, Np);
    return dst;
  };

  // ---- bf16 weight repack (small; redone per call, deterministic) ----
  bf16_t* Winb = convW(P(PB + 0), 32, 300, 32, HP);
  bf16_t *Wne[5], *Wee[5], *Wl1[5], *Wl2[5], *Wat1[5];
  for (int l = 0; l < 5; ++l) {
    Wne[l]  = convW(P(GIN(l, 0)), 300, 300, HP, HP);
    Wee[l]  = convW(P(GIN(l, 4)), 32, 300, 32, HP);
    Wl1[l]  = convW(P(GIN(l, 8)), 300, 600, HP, H2P);
    Wl2[l]  = convW(P(GIN(l, 12)), 600, 300, H2P, HP);
    Wat1[l] = convCat(P(GIN(l, 14)), 300, HP, 300, 300, H2P, HP);
  }
  bf16_t* Wvl1 = convW(P(VNB + 1), 300, 600, HP, H2P);
  bf16_t* Wvl2 = convW(P(VNB + 5), 600, 300, H2P, HP);
  bf16_t* Wva1 = convCat(P(VNB + 7), 300, HP, 300, 300, H2P, HP);
  bf16_t* Wc1  = convW(P(CLFB + 0), 900, 300, P3P, HP);
  bf16_t* Wc2  = convW(P(CLFB + 4), 300, 150, HP, 192);
  bf16_t* Wfc1 = convW(P(NFB + 0), 900, 450, P3P, 512);
  bf16_t* Wfc2 = convW(P(NFB + 4), 450, 225, 480, 256);
  bf16_t* Wfn1 = convW(P(NFB + 6), 900, 450, P3P, 512);
  bf16_t* Wfn2 = convW(P(NFB + 10), 450, 225, 480, 256);
  bf16_t* Wff1 = convW(P(NFB + 12), 450, 225, 480, 256);

  // ---- activations / scratch ----
  bf16_t* xb   = (bf16_t*)alloc((size_t)NNP * 32 * 2);
  bf16_t* eab  = (bf16_t*)alloc((size_t)NEDP * 32 * 2);
  float*  h    = (float*)alloc((size_t)NNP * HP * 4);
  bf16_t* hb   = (bf16_t*)alloc((size_t)NNP * HP * 2);
  float*  xt   = (float*)alloc((size_t)NNP * HP * 4);
  bf16_t* xtb  = (bf16_t*)alloc((size_t)NNP * HP * 2);
  bf16_t* cmb  = (bf16_t*)alloc((size_t)NNP * HP * 2);
  bf16_t* tnb  = (bf16_t*)alloc((size_t)NNP * H2P * 2);
  bf16_t* efb  = (bf16_t*)alloc((size_t)NEDP * HP * 2);
  bf16_t* t1b  = (bf16_t*)alloc((size_t)NEDP * HP * 2);
  float*  eatt = (float*)alloc((size_t)NEDP * 4);
  float*  natt = (float*)alloc((size_t)NNP * 4);
  float*  agg  = (float*)alloc((size_t)NNP * HP * 4);
  float*  gsum = (float*)alloc((size_t)NG * HP * 4);
  bf16_t* vinb = (bf16_t*)alloc((size_t)NG * HP * 2);
  bf16_t* vt1b = (bf16_t*)alloc((size_t)NG * H2P * 2);
  float*  vu   = (float*)alloc((size_t)NG * HP * 4);
  bf16_t* vub  = (bf16_t*)alloc((size_t)NG * HP * 2);
  float*  cnt  = (float*)alloc(NG * 4);
  float*  scrE = (float*)alloc(64);
  float*  scrN = (float*)alloc(64);
  float*  ps   = (float*)alloc((size_t)NG * HP * 4);
  float*  pmx  = (float*)alloc((size_t)NG * HP * 4);
  bf16_t* gbuf = (bf16_t*)alloc((size_t)NG * P3P * 2);
  bf16_t* z1b  = (bf16_t*)alloc((size_t)NG * HP * 2);
  float*  z2f  = (float*)alloc((size_t)NG * 160 * 4);
  bf16_t* nf1b = (bf16_t*)alloc((size_t)NG * 480 * 2);
  bf16_t* cmbn = (bf16_t*)alloc((size_t)NG * 480 * 2);
  float*  z3f  = (float*)alloc((size_t)NG * 256 * 4);

  auto gemm = [&](int mode, GemmA A, const bf16_t* Wt, int Kp, int Npad, int Mp, GemmEpi E) {
    dim3 g(Npad / 64, Mp / 128);
    if (mode == 0)      k_gemm<0><<<g, 256, 0, stream>>>(A, Wt, Kp, E);
    else if (mode == 1) k_gemm<1><<<g, 256, 0, stream>>>(A, Wt, Kp, E);
    else                k_gemm<2><<<g, 256, 0, stream>>>(A, Wt, Kp, E);
  };
  auto mkA = [&](const bf16_t* a, int lda, int Mreal) {
    GemmA A; A.a0 = a; A.addb = nullptr; A.idx0 = nullptr;
    A.a1 = a; A.idx1 = nullptr; A.lda0 = lda; A.lda1 = lda;
    A.ksplit = 1 << 30; A.Mreal = Mreal; return A;
  };

  // input conversions + graph counts
  k_convert_pad<<<(NNP * 32 + 255) / 256, 256, 0, stream>>>(x, xb, NN, 32, NNP, 32);
  k_convert_pad<<<(NEDP * 32 + 255) / 256, 256, 0, stream>>>(eain, eab, NED, 32, NEDP, 32);
  k_zero_f<<<(NG + 255) / 256, 256, 0, stream>>>(cnt, NG);
  k_count<<<(NN + 255) / 256, 256, 0, stream>>>(batch, cnt);

  // input encoder: h = relu(bn(x @ Win + b))
  {
    GemmEpi E{P(PB + 1), P(PB + 2), P(PB + 3), nullptr, h, hb, HP, HP, 300, HP, FLAG_RELU};
    gemm(0, mkA(xb, 32, NN), Winb, 32, HP, NNP, E);
  }

  for (int l = 0; l < 5; ++l) {
    // --- virtual node: mean pool -> MLP ---
    k_zero_f<<<(NG * HP + 255) / 256, 256, 0, stream>>>(gsum, NG * HP);
    k_segsum<<<(NN * HP + 255) / 256, 256, 0, stream>>>(h, batch, gsum);
    k_vn_prep<<<(NG * HP + 255) / 256, 256, 0, stream>>>(gsum, cnt, P(VNB + 0), vinb);
    { GemmEpi E{P(VNB + 2), P(VNB + 3), P(VNB + 4), nullptr, nullptr, vt1b, H2P, H2P, 600, H2P, FLAG_RELU};
      gemm(0, mkA(vinb, HP, NG), Wvl1, HP, H2P, NG, E); }
    { GemmEpi E{P(VNB + 6), nullptr, nullptr, nullptr, vu, vub, HP, HP, 300, HP, 0};
      gemm(0, mkA(vt1b, H2P, NG), Wvl2, H2P, HP, NG, E); }
    // vn attention: tanh(a1([h, vn_upd[batch]])) -> a2 -> global softmax
    { GemmA A; A.a0 = hb; A.addb = nullptr; A.idx0 = nullptr;
      A.a1 = vub; A.idx1 = batch; A.lda0 = HP; A.lda1 = HP; A.ksplit = HP; A.Mreal = NN;
      GemmEpi E{P(VNB + 8), nullptr, nullptr, nullptr, nullptr, tnb, HP, HP, 300, HP, FLAG_TANH};
      gemm(1, A, Wva1, H2P, HP, NNP, E); }
    k_dot_reduce<<<NNP / 8, 256, 0, stream>>>(tnb, HP, 300, P(VNB + 9), P(VNB + 10), 1.0f, NN, natt);
    k_scr_init<<<1, 32, 0, stream>>>(scrN);
    k_amax<<<(NNP + 255) / 256, 256, 0, stream>>>(natt, NNP, scrN);
    k_expsum<<<(NNP + 255) / 256, 256, 0, stream>>>(natt, NNP, scrN);

    // --- GIN+ layer ---
    { GemmEpi E{P(GIN(l, 1)), P(GIN(l, 2)), P(GIN(l, 3)), nullptr, xt, xtb, HP, HP, 300, HP, FLAG_RELU};
      gemm(0, mkA(hb, HP, NN), Wne[l], HP, HP, NNP, E); }
    { GemmEpi E{P(GIN(l, 5)), P(GIN(l, 6)), P(GIN(l, 7)), nullptr, nullptr, efb, HP, HP, 300, HP, FLAG_RELU};
      gemm(0, mkA(eab, 32, NED), Wee[l], 32, HP, NEDP, E); }
    // edge attention: tanh(att1([xt[row]+ef, xt[col]])) ; gathers fused in A-load
    { GemmA A; A.a0 = xtb; A.addb = efb; A.idx0 = row;
      A.a1 = xtb; A.idx1 = col; A.lda0 = HP; A.lda1 = HP; A.ksplit = HP; A.Mreal = NED;
      GemmEpi E{P(GIN(l, 15)), nullptr, nullptr, nullptr, nullptr, t1b, HP, HP, 300, HP, FLAG_TANH};
      gemm(2, A, Wat1[l], H2P, HP, NEDP, E); }
    k_dot_reduce<<<NEDP / 8, 256, 0, stream>>>(t1b, HP, 300, P(GIN(l, 16)), P(GIN(l, 17)), 0.2f, NED, eatt);
    k_scr_init<<<1, 32, 0, stream>>>(scrE);
    k_amax<<<(NEDP + 255) / 256, 256, 0, stream>>>(eatt, NEDP, scrE);
    k_expsum<<<(NEDP + 255) / 256, 256, 0, stream>>>(eatt, NEDP, scrE);
    // scatter-add aggregation
    k_zero_f<<<(NNP * HP + 255) / 256, 256, 0, stream>>>(agg, NNP * HP);
    k_scatter<<<NEDP / 8, 256, 0, stream>>>(xtb, efb, row, col, eatt, scrE, agg);
    // out = (1+eps)*xt + agg + vexp*vaw
    k_combine<<<(NNP * HP + 255) / 256, 256, 0, stream>>>(xt, P(GIN(l, 18)), agg, vu, batch,
                                                          natt, scrN, cmb);
    // node MLP (+ residual for l>0 fused into epilogue)
    { GemmEpi E{P(GIN(l, 9)), P(GIN(l, 10)), P(GIN(l, 11)), nullptr, nullptr, tnb, H2P, H2P, 600, H2P, FLAG_RELU};
      gemm(0, mkA(cmb, HP, NN), Wl1[l], HP, H2P, NNP, E); }
    { GemmEpi E{P(GIN(l, 13)), nullptr, nullptr, (l > 0 ? h : nullptr), h, hb, HP, HP, 300, HP, 0};
      gemm(0, mkA(tnb, H2P, NN), Wl2[l], H2P, HP, NNP, E); }
  }

  // --- pooling (sum / mean / max) ---
  float* logits = (float*)d_out;
  float* conf   = logits + NG * 6;
  float* gout   = conf + NG;
  k_zero_f<<<(NG * HP + 255) / 256, 256, 0, stream>>>(ps, NG * HP);
  k_fill_f<<<(NG * HP + 255) / 256, 256, 0, stream>>>(pmx, NG * HP, -1e30f);
  k_pool_acc<<<(NN * HP + 255) / 256, 256, 0, stream>>>(h, batch, ps, pmx);
  k_pool_pack<<<(NG * P3P + 255) / 256, 256, 0, stream>>>(ps, pmx, cnt, gbuf, gout);

  // --- classifier ---
  { GemmEpi E{P(CLFB + 1), P(CLFB + 2), P(CLFB + 3), nullptr, nullptr, z1b, HP, HP, 300, HP, FLAG_RELU};
    gemm(0, mkA(gbuf, P3P, NG), Wc1, P3P, HP, NG, E); }
  { GemmEpi E{P(CLFB + 5), P(CLFB + 6), P(CLFB + 7), nullptr, z2f, nullptr, 160, 160, 150, 160, FLAG_RELU};
    gemm(0, mkA(z1b, HP, NG), Wc2, HP, 192, NG, E); }
  k_l3<<<(NG * 6 + 255) / 256, 256, 0, stream>>>(z2f, P(CLFB + 8), P(CLFB + 9), logits);

  // --- NCOD noise filter ---
  k_zero_b<<<(NG * 480 + 255) / 256, 256, 0, stream>>>(cmbn, NG * 480);
  { GemmEpi E{P(NFB + 1), P(NFB + 2), P(NFB + 3), nullptr, nullptr, nf1b, 480, 480, 450, 480, FLAG_RELU};
    gemm(0, mkA(gbuf, P3P, NG), Wfc1, P3P, 512, NG, E); }
  { GemmEpi E{P(NFB + 5), nullptr, nullptr, nullptr, nullptr, cmbn, 480, 480, 225, 225, 0};
    gemm(0, mkA(nf1b, 480, NG), Wfc2, 480, 256, NG, E); }
  { GemmEpi E{P(NFB + 7), P(NFB + 8), P(NFB + 9), nullptr, nullptr, nf1b, 480, 480, 450, 480, FLAG_RELU};
    gemm(0, mkA(gbuf, P3P, NG), Wfn1, P3P, 512, NG, E); }
  { GemmEpi E{P(NFB + 11), nullptr, nullptr, nullptr, nullptr, cmbn + 225, 480, 480, 225, 225, 0};
    gemm(0, mkA(nf1b, 480, NG), Wfn2, 480, 256, NG, E); }
  { GemmEpi E{P(NFB + 13), nullptr, nullptr, nullptr, z3f, nullptr, 256, 256, 225, 256, FLAG_RELU};
    gemm(0, mkA(cmbn, 480, NG), Wff1, 480, 256, NG, E); }
  k_f2<<<NG / 8, 256, 0, stream>>>(z3f, P(NFB + 14), P(NFB + 15), conf);
}
